// GAT_996432413503
// MI455X (gfx1250) — matrix operand order repaired and checked by reading, not compiled, not run on hardware
//
#include <hip/hip_runtime.h>
#include <hip/hip_bf16.h>
#include <math.h>

typedef __attribute__((ext_vector_type(16))) _Float16 v16h;
typedef __attribute__((ext_vector_type(8)))  float    v8f;

#define HEADS 4
#define CH 32
#define DF 128
#define NGR 64
#define NEG_SLOPE 0.2f

// ---------------------------------------------------------------- utilities

__global__ void fill_kernel(float* __restrict__ p, long n, float v) {
  long i = (long)blockIdx.x * blockDim.x + threadIdx.x;
  if (i < n) p[i] = v;
}

__device__ __forceinline__ float leakyf(float x) {
  return x > 0.f ? x : NEG_SLOPE * x;
}

// float atomic max via ordered-int trick (init with -inf)
__device__ __forceinline__ void atomicMaxF(float* addr, float v) {
  if (v >= 0.f) atomicMax((int*)addr, __float_as_int(v));
  else          atomicMin((unsigned int*)addr, (unsigned int)__float_as_int(v));
}

// ------------------------------------------------- xw = in[N,128] @ W[128,128]
// One wave computes a 16x16 tile of xw via v_wmma_f32_16x16x32_f16 (4 K-steps).
// blockDim = 256 (8 waves) -> one block covers 16 rows x all 128 cols.
// A layout (16-bit A 16x32, ISA 7.12.2): lane<16 holds row m=lane, K in
//   {k0..k0+7, k0+16..k0+23} with k0 = kb + 8*(lane>=16).
// B layout: same K order as A, lane group g holds K = kb+8g+{0..7, 16..23}.
// C/D layout: vgpr r, lane L -> M = r + 8*(L>>4), N = L&15.
__global__ __launch_bounds__(256)
void gemm_xw_kernel(const float* __restrict__ in, const float* __restrict__ W,
                    float* __restrict__ xw, int N) {
  int wave  = threadIdx.x >> 5;
  int lane  = threadIdx.x & 31;
  int row0  = blockIdx.x << 4;
  int col0  = wave << 4;
  int m     = lane & 15;
  int khalf = lane >> 4;
  int col   = col0 + m;

  const float* arow = in + (size_t)(row0 + m) * DF;
  v8f acc = {};
  #pragma unroll
  for (int kb = 0; kb < DF; kb += 32) {
    int k0a = kb + 8  * khalf;  // A-matrix K base for this lane half
    int k0b = kb + 8  * khalf;  // B-matrix K base for this lane half
    v16h a, b;
    #pragma unroll
    for (int e = 0; e < 8; ++e) {
      a[e]     = (_Float16)arow[k0a + e];
      a[e + 8] = (_Float16)arow[k0a + e + 16];
    }
    #pragma unroll
    for (int e = 0; e < 16; ++e) {
      b[e] = (_Float16)W[(size_t)(k0b + (e & 7) + ((e >> 3) << 4)) * DF + col];
    }
    acc = __builtin_amdgcn_wmma_f32_16x16x32_f16(
        /*neg_a=*/false, a, /*neg_b=*/false, b,
        /*c_mod=*/(short)0, acc, /*reuse_a=*/false, /*reuse_b=*/false);
  }
  #pragma unroll
  for (int r = 0; r < 8; ++r) {
    int M = r + 8 * khalf;
    xw[(size_t)(row0 + M) * DF + col] = acc[r];
  }
}

// ------------------------------------------------- al/ar = einsum(nhc,hc->nh)
__global__ void attn_dots_kernel(const float* __restrict__ xw,
                                 const float* __restrict__ attl,
                                 const float* __restrict__ attr,
                                 float* __restrict__ al, float* __restrict__ ar,
                                 int N) {
  int i = blockIdx.x * blockDim.x + threadIdx.x;
  if (i >= N * HEADS) return;
  int node = i >> 2, h = i & 3;
  const float* v = xw + (size_t)node * DF + h * CH;
  float sl = 0.f, sr = 0.f;
  #pragma unroll
  for (int c = 0; c < CH; ++c) {
    sl += v[c] * attl[h * CH + c];
    sr += v[c] * attr[h * CH + c];
  }
  al[i] = sl;
  ar[i] = sr;
}

// ------------------------------------------------- edge passes (E real + N self-loops)
__global__ void edge_max_kernel(const int* __restrict__ src, const int* __restrict__ dst,
                                const float* __restrict__ al, const float* __restrict__ ar,
                                float* __restrict__ mb, int E, int N) {
  long i = (long)blockIdx.x * blockDim.x + threadIdx.x;
  long total = (long)(E + N) * HEADS;
  if (i >= total) return;
  int h = (int)(i & 3);
  long e = i >> 2;
  int s, d;
  if (e < E) { s = src[e]; d = dst[e]; } else { s = d = (int)(e - E); }
  float l = leakyf(al[(size_t)s * HEADS + h] + ar[(size_t)d * HEADS + h]);
  atomicMaxF(&mb[(size_t)d * HEADS + h], l);
}

__global__ void edge_sum_kernel(const int* __restrict__ src, const int* __restrict__ dst,
                                const float* __restrict__ al, const float* __restrict__ ar,
                                const float* __restrict__ mb, float* __restrict__ zb,
                                int E, int N) {
  long i = (long)blockIdx.x * blockDim.x + threadIdx.x;
  long total = (long)(E + N) * HEADS;
  if (i >= total) return;
  int h = (int)(i & 3);
  long e = i >> 2;
  int s, d;
  if (e < E) { s = src[e]; d = dst[e]; } else { s = d = (int)(e - E); }
  size_t dh = (size_t)d * HEADS + h;
  float l = leakyf(al[(size_t)s * HEADS + h] + ar[dh]);
  atomicAdd(&zb[dh], __expf(l - mb[dh]));
}

// one thread per (edge, channel): 128 contiguous atomic adds per edge row
__global__ void edge_aggr_kernel(const int* __restrict__ src, const int* __restrict__ dst,
                                 const float* __restrict__ al, const float* __restrict__ ar,
                                 const float* __restrict__ mb, const float* __restrict__ zb,
                                 const float* __restrict__ xw, float* __restrict__ out,
                                 int E, int N) {
  long i = (long)blockIdx.x * blockDim.x + threadIdx.x;
  long total = (long)(E + N) * DF;
  if (i >= total) return;
  int c = (int)(i & (DF - 1));
  long e = i >> 7;
  int h = c >> 5;
  int s, d;
  if (e < E) { s = src[e]; d = dst[e]; } else { s = d = (int)(e - E); }
  size_t dh = (size_t)d * HEADS + h;
  float l = leakyf(al[(size_t)s * HEADS + h] + ar[dh]);
  float alpha = __expf(l - mb[dh]) / zb[dh];
  atomicAdd(&out[(size_t)d * DF + c], alpha * xw[(size_t)s * DF + c]);
}

__global__ void bias_relu_kernel(float* __restrict__ h, const float* __restrict__ bias,
                                 long n) {
  long i = (long)blockIdx.x * blockDim.x + threadIdx.x;
  if (i >= n) return;
  int c = (int)(i & (DF - 1));
  h[i] = fmaxf(h[i] + bias[c], 0.f);
}

// ------------------------------------------------- global mean pool + MLP head
__global__ void pool_kernel(const float* __restrict__ h, const int* __restrict__ batch,
                            float* __restrict__ pooled, float* __restrict__ cnt, int N) {
  long i = (long)blockIdx.x * blockDim.x + threadIdx.x;
  long total = (long)N * DF;
  if (i >= total) return;
  int c = (int)(i & (DF - 1));
  int node = (int)(i >> 7);
  int b = batch[node];
  atomicAdd(&pooled[(size_t)b * DF + c], h[i]);
  if (c == 0) atomicAdd(&cnt[b], 1.f);
}

__global__ void mlp_head_kernel(const float* __restrict__ pooled, const float* __restrict__ cnt,
                                const float* __restrict__ lin1_w, const float* __restrict__ lin1_b,
                                const float* __restrict__ lin2_w, const float* __restrict__ lin2_b,
                                float* __restrict__ out) {
  __shared__ float p[DF];
  __shared__ float gv[DF];
  __shared__ float lg[10];
  int g = blockIdx.x, t = threadIdx.x;
  float denom = fmaxf(cnt[g], 1.f);
  p[t] = pooled[(size_t)g * DF + t] / denom;
  __syncthreads();
  float s = lin1_b[t];
  for (int k = 0; k < DF; ++k) s += p[k] * lin1_w[k * DF + t];
  gv[t] = fmaxf(s, 0.f);
  __syncthreads();
  if (t < 10) {
    float s2 = lin2_b[t];
    for (int k = 0; k < DF; ++k) s2 += gv[k] * lin2_w[k * 10 + t];
    lg[t] = s2;
  }
  __syncthreads();
  if (t == 0) {
    float mx = lg[0];
    for (int j = 1; j < 10; ++j) mx = fmaxf(mx, lg[j]);
    float ss = 0.f;
    for (int j = 0; j < 10; ++j) ss += __expf(lg[j] - mx);
    float lse = mx + __logf(ss);
    for (int j = 0; j < 10; ++j) out[(size_t)g * 10 + j] = lg[j] - lse;
  }
}

// ---------------------------------------------------------------- launcher

extern "C" void kernel_launch(void* const* d_in, const int* in_sizes, int n_in,
                              void* d_out, int out_size, void* d_ws, size_t ws_size,
                              hipStream_t stream) {
  (void)n_in; (void)out_size; (void)ws_size;

  const float* x     = (const float*)d_in[0];
  const int*   ei    = (const int*)d_in[1];
  const int*   batch = (const int*)d_in[2];
  const float* Wl[3] = {(const float*)d_in[3], (const float*)d_in[7],  (const float*)d_in[11]};
  const float* AL[3] = {(const float*)d_in[4], (const float*)d_in[8],  (const float*)d_in[12]};
  const float* AR[3] = {(const float*)d_in[5], (const float*)d_in[9],  (const float*)d_in[13]};
  const float* Bl[3] = {(const float*)d_in[6], (const float*)d_in[10], (const float*)d_in[14]};
  const float* lin1_w = (const float*)d_in[15];
  const float* lin1_b = (const float*)d_in[16];
  const float* lin2_w = (const float*)d_in[17];
  const float* lin2_b = (const float*)d_in[18];

  const int N = in_sizes[0] / DF;       // 100000
  const int E = in_sizes[1] / 2;        // 1600000
  const int* src = ei;
  const int* dst = ei + E;

  float* out        = (float*)d_out;
  float* logits_out = out;              // [64,10]
  float* hLast      = out + NGR * 10;   // [N,128] ("last" output, also activation ping-pong)

  // workspace layout (floats)
  float* ws     = (float*)d_ws;
  size_t NN     = (size_t)N * DF;
  float* xw     = ws;                               // N*128
  float* hA     = xw + NN;                          // N*128
  float* al     = hA + NN;                          // N*4
  float* ar     = al + (size_t)N * HEADS;           // N*4
  float* mb     = ar + (size_t)N * HEADS;           // N*4
  float* zb     = mb + (size_t)N * HEADS;           // N*4
  float* pooled = zb + (size_t)N * HEADS;           // 64*128
  float* cnt    = pooled + NGR * DF;                // 64 (contiguous with pooled)

  const float* ins[3]  = {x, hLast, hA};
  float*       outs[3] = {hLast, hA, hLast};

  const long teh = (long)(E + N) * HEADS;
  const long tec = (long)(E + N) * DF;

  for (int L = 0; L < 3; ++L) {
    gemm_xw_kernel<<<N / 16, 256, 0, stream>>>(ins[L], Wl[L], xw, N);
    attn_dots_kernel<<<(N * HEADS + 255) / 256, 256, 0, stream>>>(xw, AL[L], AR[L], al, ar, N);
    fill_kernel<<<((long)N * HEADS + 255) / 256, 256, 0, stream>>>(mb, (long)N * HEADS,
                                                                   -__builtin_huge_valf());
    fill_kernel<<<((long)N * HEADS + 255) / 256, 256, 0, stream>>>(zb, (long)N * HEADS, 0.f);
    fill_kernel<<<(long)((NN + 255) / 256), 256, 0, stream>>>(outs[L], (long)NN, 0.f);
    edge_max_kernel<<<(teh + 255) / 256, 256, 0, stream>>>(src, dst, al, ar, mb, E, N);
    edge_sum_kernel<<<(teh + 255) / 256, 256, 0, stream>>>(src, dst, al, ar, mb, zb, E, N);
    edge_aggr_kernel<<<(tec + 255) / 256, 256, 0, stream>>>(src, dst, al, ar, mb, zb, xw,
                                                            outs[L], E, N);
    bias_relu_kernel<<<(long)((NN + 255) / 256), 256, 0, stream>>>(outs[L], Bl[L], (long)NN);
  }

  // global mean pool + head
  fill_kernel<<<(NGR * DF + NGR + 255) / 256, 256, 0, stream>>>(pooled,
                                                                (long)(NGR * DF + NGR), 0.f);
  pool_kernel<<<(long)((NN + 255) / 256), 256, 0, stream>>>(hLast, batch, pooled, cnt, N);
  mlp_head_kernel<<<NGR, DF, 0, stream>>>(pooled, cnt, lin1_w, lin1_b, lin2_w, lin2_b,
                                          logits_out);
}
